// LSTMVAEEncoder_56032143343745
// MI455X (gfx1250) — matrix operand order repaired and checked by reading, not compiled, not run on hardware
//
#include <hip/hip_runtime.h>
#include <hip/hip_bf16.h>
#include <stdint.h>

// ---------------------------------------------------------------------------
// LSTM VAE encoder for MI455X (gfx1250)
//   - v_wmma_f32_16x16x32_bf16 for both GEMM paths (fp32 accumulate)
//   - global_load_async_to_lds_b128 double-buffered B-tile staging (ASYNCcnt)
//   - A-fragment prefetch placed AFTER the visibility barrier so it resolves
//     at the post-WMMA barrier (overlaps the matrix pipe)
//   - K-loop unrolled 2x so the LDS double-buffer index is compile-time
// B=64, T=256, E=H=512, G=4H=2048, L=8
// ---------------------------------------------------------------------------

typedef __attribute__((ext_vector_type(16))) __bf16        v16bf;
typedef __attribute__((ext_vector_type(8)))  float         v8f;
typedef __attribute__((ext_vector_type(4)))  unsigned int  v4u;

#define B_ 64
#define T_ 256
#define H_ 512
#define G_ 2048
#define L_ 8
#define KSTEP 32
#define BN 64      // block N tile (shared B slice)
#define BM 128     // block M tile (4 waves x 32 rows)

// ---- helpers --------------------------------------------------------------

__device__ __forceinline__ unsigned short f2bf(float f) {
    unsigned int u = __float_as_uint(f);
    unsigned int r = u + 0x7FFFu + ((u >> 16) & 1u);   // round-to-nearest-even
    return (unsigned short)(r >> 16);
}

union Frag { v4u u[2]; v16bf v; };

// two aligned 16B chunks -> 16 bf16 fragment (works for global or LDS ptrs)
__device__ __forceinline__ v16bf load_frag2(const unsigned short* p0,
                                            const unsigned short* p1) {
    Frag f;
    f.u[0] = *(const v4u*)p0;
    f.u[1] = *(const v4u*)p1;
    return f.v;
}

__device__ __forceinline__ float sigmoidf_(float x) {
    return 1.0f / (1.0f + expf(-x));
}

// async copy 16B global -> LDS, tracked by ASYNCcnt
__device__ __forceinline__ void async_copy16(unsigned lds_addr, const void* gptr) {
    asm volatile("global_load_async_to_lds_b128 %0, %1, off"
                 :: "v"(lds_addr), "v"(gptr)
                 : "memory");
}

// ---- fp32 -> bf16 conversion ---------------------------------------------

__global__ void f32_to_bf16_kernel(const float* __restrict__ src,
                                   unsigned short* __restrict__ dst, int n) {
    int i = blockIdx.x * blockDim.x + threadIdx.x;
    int stride = gridDim.x * blockDim.x;
    for (; i < n; i += stride) dst[i] = f2bf(src[i]);
}

// ---- pre = seq @ W_ih^T + b_ih + b_hh ------------------------------------
// A: (16384 x 512) bf16 row-major, W: (2048 x 512) bf16 row-major (B = W^T).
// Block = 128 threads (4 waves). Block tile M=128 (wave: 32 rows), N=64.
// B tile (64 rows x 32 k = 4KB) staged in LDS via async copies, double buffer.
__global__ void __launch_bounds__(128)
gemm_pre_kernel(const unsigned short* __restrict__ A,
                const unsigned short* __restrict__ W,
                const float* __restrict__ bih,
                const float* __restrict__ bhh,
                float* __restrict__ pre) {
    __shared__ __align__(16) unsigned short btile[2][BN * KSTEP];  // 2 x 4KB

    const int tid  = threadIdx.x;
    const int lane = tid & 31;
    const int wave = tid >> 5;
    const int lo = lane & 15, hi = lane >> 4;
    const int mbase = blockIdx.y * BM + wave * 32;
    const int nbase = blockIdx.x * BN;

    const unsigned lds0 = (unsigned)(unsigned long long)(&btile[0][0]);
    const unsigned lds1 = (unsigned)(unsigned long long)(&btile[1][0]);

    // issue async copies of W[nbase..nbase+63][k..k+31] into LDS buffer
    auto issue_b = [&](int buf, int k) {
        unsigned base = buf ? lds1 : lds0;
        // 4KB tile = 256 x 16B chunks, 128 threads -> 2 chunks each
#pragma unroll
        for (int i = 0; i < 2; ++i) {
            int c = tid + i * 128;
            int row = c >> 2;                 // 0..63
            int koff = (c & 3) * 8;           // element offset in k slice
            const unsigned short* g = W + (size_t)(nbase + row) * H_ + k + koff;
            async_copy16(base + (unsigned)c * 16u, (const void*)g);
        }
    };

    issue_b(0, 0);
    issue_b(1, KSTEP);

    const unsigned short* arow0 = A + (size_t)(mbase + lo) * H_;
    const unsigned short* arow1 = A + (size_t)(mbase + 16 + lo) * H_;

    v8f acc0[4], acc1[4];
#pragma unroll
    for (int g = 0; g < 4; ++g)
#pragma unroll
        for (int r = 0; r < 8; ++r) { acc0[g][r] = 0.0f; acc1[g][r] = 0.0f; }

    v16bf a0 = load_frag2(arow0 + hi * 8, arow0 + hi * 8 + 16);
    v16bf a1 = load_frag2(arow1 + hi * 8, arow1 + hi * 8 + 16);

    const int NK = H_ / KSTEP;   // 16

    // one K-step; buf is a compile-time constant (loop below unrolled by 2)
    auto kstep_body = [&](int kk, int buf) {
        // current buffer's copies were issued 2 iterations ago; <=2 younger
        // copies may remain outstanding (async loads complete in order).
        if (kk + 1 < NK) asm volatile("s_wait_asynccnt 0x2" ::: "memory");
        else             asm volatile("s_wait_asynccnt 0x0" ::: "memory");
        __syncthreads();   // B tile visible to all waves

        // prefetch next A fragments AFTER the barrier: their loadcnt wait
        // lands at the post-WMMA barrier, hidden behind the matrix pipe.
        v16bf a0n = a0, a1n = a1;
        if (kk + 1 < NK) {
            int kn = (kk + 1) * KSTEP;
            a0n = load_frag2(arow0 + kn + hi * 8, arow0 + kn + hi * 8 + 16);
            a1n = load_frag2(arow1 + kn + hi * 8, arow1 + kn + hi * 8 + 16);
        }

        const unsigned short* bbase = &btile[buf][0];
        v16bf bf[4];
#pragma unroll
        for (int g = 0; g < 4; ++g) {
            const unsigned short* p = bbase + (g * 16 + lo) * KSTEP + hi * 8;
            bf[g] = load_frag2(p, p + 16);
        }
#pragma unroll
        for (int g = 0; g < 4; ++g) {
            acc0[g] = __builtin_amdgcn_wmma_f32_16x16x32_bf16(
                false, a0, false, bf[g], (short)0, acc0[g], false, false);
            acc1[g] = __builtin_amdgcn_wmma_f32_16x16x32_bf16(
                false, a1, false, bf[g], (short)0, acc1[g], false, false);
        }
        __syncthreads();   // everyone done reading buf before refilling it
        if (kk + 2 < NK) issue_b(buf, (kk + 2) * KSTEP);
        a0 = a0n; a1 = a1n;
    };

#pragma unroll 1
    for (int kk = 0; kk < NK; kk += 2) {
        kstep_body(kk, 0);
        kstep_body(kk + 1, 1);
    }

#pragma unroll
    for (int g = 0; g < 4; ++g) {
        int n = nbase + g * 16 + lo;
        float bias = bih[n] + bhh[n];
#pragma unroll
        for (int r = 0; r < 8; ++r) {
            int m0 = mbase + r + 8 * hi;
            pre[(size_t)m0 * G_ + n] = acc0[g][r] + bias;
            pre[(size_t)(m0 + 16) * G_ + n] = acc1[g][r] + bias;
        }
    }
}

// ---- one fused LSTM timestep ---------------------------------------------
// g = pre[:,t,:] + h(t-1) @ Whh^T ; gate math ; write c, h(bf16 seq), finals.
// grid = 32 blocks (j tiles of 16), block = 128 threads: wave -> b-tile of 16.
// Each wave computes the SAME j block for all 4 gates (A fragment reused 4x).
// Latency-critical: direct global loads (Whh is L2-resident, 2MB/layer),
// register double-buffered so WMMAs overlap the next K-step's loads.
__global__ void __launch_bounds__(128)
lstm_step_kernel(const float* __restrict__ pre,
                 const unsigned short* __restrict__ Whh,
                 unsigned short* __restrict__ seq,   // (B*T x 512) bf16, rd t-1 / wr t
                 float* __restrict__ c,              // (B x 512)
                 float* __restrict__ finalsl,        // + l*H_, row stride L_*H_
                 int t) {
    const int lane = threadIdx.x & 31;
    const int wave = threadIdx.x >> 5;
    const int lo = lane & 15, hi = lane >> 4;
    const int mbase = wave * 16;         // batch tile
    const int jbase = blockIdx.x * 16;   // hidden tile

    v8f acc[4];
#pragma unroll
    for (int g = 0; g < 4; ++g) {
        int n = g * H_ + jbase + lo;
#pragma unroll
        for (int r = 0; r < 8; ++r) {
            int b = mbase + r + 8 * hi;
            acc[g][r] = pre[((size_t)b * T_ + t) * G_ + n];
        }
    }

    if (t > 0) {
        const unsigned short* arow =
            seq + ((size_t)(mbase + lo) * T_ + (t - 1)) * H_;
        const unsigned short* wr[4];
#pragma unroll
        for (int g = 0; g < 4; ++g)
            wr[g] = Whh + (size_t)(g * H_ + jbase + lo) * H_;

        v16bf af = load_frag2(arow + hi * 8, arow + hi * 8 + 16);
        v16bf bf[4];
#pragma unroll
        for (int g = 0; g < 4; ++g)
            bf[g] = load_frag2(wr[g] + hi * 8, wr[g] + hi * 8 + 16);

#pragma unroll 1
        for (int k = 0; k < H_ - KSTEP; k += KSTEP) {
            const int kn = k + KSTEP;
            // prefetch next fragments before consuming current ones
            v16bf afn = load_frag2(arow + kn + hi * 8, arow + kn + hi * 8 + 16);
            v16bf bfn[4];
#pragma unroll
            for (int g = 0; g < 4; ++g)
                bfn[g] = load_frag2(wr[g] + kn + hi * 8, wr[g] + kn + hi * 8 + 16);
#pragma unroll
            for (int g = 0; g < 4; ++g)
                acc[g] = __builtin_amdgcn_wmma_f32_16x16x32_bf16(
                    false, af, false, bf[g], (short)0, acc[g], false, false);
            af = afn;
#pragma unroll
            for (int g = 0; g < 4; ++g) bf[g] = bfn[g];
        }
#pragma unroll
        for (int g = 0; g < 4; ++g)
            acc[g] = __builtin_amdgcn_wmma_f32_16x16x32_bf16(
                false, af, false, bf[g], (short)0, acc[g], false, false);
    }

#pragma unroll
    for (int r = 0; r < 8; ++r) {
        int b = mbase + r + 8 * hi;
        int j = jbase + lo;
        float cp = (t > 0) ? c[(size_t)b * H_ + j] : 0.0f;
        float iv = sigmoidf_(acc[0][r]);
        float fv = sigmoidf_(acc[1][r]);
        float gv = tanhf(acc[2][r]);
        float ov = sigmoidf_(acc[3][r]);
        float cn = fv * cp + iv * gv;
        float hv = ov * tanhf(cn);
        c[(size_t)b * H_ + j] = cn;
        seq[((size_t)b * T_ + t) * H_ + j] = f2bf(hv);
        if (t == T_ - 1) finalsl[(size_t)b * (L_ * H_) + j] = hv;
    }
}

// ---- head: ff1 (silu), ff2, mean/scale split (fp32, tiny) ----------------

__global__ void ff1_kernel(const float* __restrict__ fin,   // 512 x 512
                           const float* __restrict__ w,     // 128 x 512
                           const float* __restrict__ bias,  // 128
                           float* __restrict__ out) {       // 512 x 128
    int idx = blockIdx.x * blockDim.x + threadIdx.x;
    if (idx >= 512 * 128) return;
    int m = idx >> 7, n = idx & 127;
    const float* a = fin + (size_t)m * 512;
    const float* ww = w + (size_t)n * 512;
    float s = bias[n];
    for (int k = 0; k < 512; ++k) s += a[k] * ww[k];
    out[idx] = s / (1.0f + expf(-s));   // silu
}

__global__ void ff2_kernel(const float* __restrict__ in,    // 512 x 128
                           const float* __restrict__ w,     // 64 x 128
                           const float* __restrict__ bias,  // 64
                           float* __restrict__ out) {       // 512 x 64
    int idx = blockIdx.x * blockDim.x + threadIdx.x;
    if (idx >= 512 * 64) return;
    int m = idx >> 6, n = idx & 63;
    const float* a = in + (size_t)m * 128;
    const float* ww = w + (size_t)n * 128;
    float s = bias[n];
    for (int k = 0; k < 128; ++k) s += a[k] * ww[k];
    out[idx] = s;
}

__global__ void head_out_kernel(const float* __restrict__ h2,  // (B,8,64) = b*512 + j
                                float* __restrict__ out) {     // mean | scale
    int idx = blockIdx.x * blockDim.x + threadIdx.x;
    if (idx >= 64 * 256) return;
    int b = idx >> 8, jj = idx & 255;
    float mean = h2[(size_t)b * 512 + 2 * jj];
    float lv   = h2[(size_t)b * 512 + 2 * jj + 1];
    float sp = (lv > 20.0f) ? lv : log1pf(expf(lv));   // softplus
    out[idx] = mean;
    out[64 * 256 + idx] = sp + 1e-8f;
}

// ---- host launch ----------------------------------------------------------

extern "C" void kernel_launch(void* const* d_in, const int* in_sizes, int n_in,
                              void* d_out, int out_size, void* d_ws, size_t ws_size,
                              hipStream_t stream) {
    (void)in_sizes; (void)n_in; (void)out_size; (void)ws_size;
    const float* x     = (const float*)d_in[0];
    const float* w_ih  = (const float*)d_in[1];
    const float* w_hh  = (const float*)d_in[2];
    const float* b_ih  = (const float*)d_in[3];
    const float* b_hh  = (const float*)d_in[4];
    const float* ff1_w = (const float*)d_in[5];
    const float* ff1_b = (const float*)d_in[6];
    const float* ff2_w = (const float*)d_in[7];
    const float* ff2_b = (const float*)d_in[8];
    float* out = (float*)d_out;

    char* ws = (char*)d_ws;
    size_t off = 0;
    auto alloc = [&](size_t bytes) -> char* {
        char* p = ws + off;
        off += (bytes + 255) & ~(size_t)255;
        return p;
    };
    unsigned short* wih16  = (unsigned short*)alloc((size_t)L_ * G_ * H_ * 2);
    unsigned short* whh16  = (unsigned short*)alloc((size_t)L_ * G_ * H_ * 2);
    unsigned short* seqA   = (unsigned short*)alloc((size_t)B_ * T_ * H_ * 2);
    unsigned short* seqB   = (unsigned short*)alloc((size_t)B_ * T_ * H_ * 2);
    float* pre    = (float*)alloc((size_t)B_ * T_ * G_ * 4);
    float* cbuf   = (float*)alloc((size_t)B_ * H_ * 4);
    float* finals = (float*)alloc((size_t)B_ * L_ * H_ * 4);
    float* f1     = (float*)alloc((size_t)512 * 128 * 4);
    float* f2     = (float*)alloc((size_t)512 * 64 * 4);

    const int nW = L_ * G_ * H_;
    f32_to_bf16_kernel<<<2048, 256, 0, stream>>>(w_ih, wih16, nW);
    f32_to_bf16_kernel<<<2048, 256, 0, stream>>>(w_hh, whh16, nW);
    f32_to_bf16_kernel<<<2048, 256, 0, stream>>>(x, seqA, B_ * T_ * H_);

    unsigned short* sin = seqA;
    unsigned short* sout = seqB;
    for (int l = 0; l < L_; ++l) {
        gemm_pre_kernel<<<dim3(G_ / BN, (B_ * T_) / BM), 128, 0, stream>>>(
            sin, wih16 + (size_t)l * G_ * H_,
            b_ih + (size_t)l * G_, b_hh + (size_t)l * G_, pre);
        for (int t = 0; t < T_; ++t) {
            lstm_step_kernel<<<H_ / 16, 128, 0, stream>>>(
                pre, whh16 + (size_t)l * G_ * H_, sout, cbuf,
                finals + (size_t)l * H_, t);
        }
        unsigned short* tmp = sin; sin = sout; sout = tmp;
    }

    ff1_kernel<<<(512 * 128 + 255) / 256, 256, 0, stream>>>(finals, ff1_w, ff1_b, f1);
    ff2_kernel<<<(512 * 64 + 255) / 256, 256, 0, stream>>>(f1, ff2_w, ff2_b, f2);
    head_out_kernel<<<(64 * 256 + 255) / 256, 256, 0, stream>>>(f2, out);
}
